// MultiHeadCovProbeV2_28879360098724
// MI455X (gfx1250) — matrix operand order repaired and checked by reading, not compiled, not run on hardware
//
#include <hip/hip_runtime.h>
#include <hip/hip_bf16.h>

typedef __attribute__((ext_vector_type(16))) _Float16 v16h;
typedef __attribute__((ext_vector_type(8)))  _Float16 v8h;
typedef __attribute__((ext_vector_type(8)))  float    v8f;
typedef __attribute__((ext_vector_type(4)))  unsigned int u32x4;
typedef __attribute__((ext_vector_type(4)))  int int32x4;
typedef __attribute__((ext_vector_type(8)))  int int32x8;

#define D_MODEL 4096
#define SEQ     2048
#define BATCH   8
#define KTILES  (D_MODEL/32)    // 128
#define EPSV    1e-3f

// proj tiling
#define KC          4                       // k-tiles per staged chunk
#define NCHUNK      (KTILES/KC)             // 32
#define CHUNK_HALVES (KC*8*32*16)           // 16384 halves = 32 KB
#define CHUNK_DW    (CHUNK_HALVES/2)        // 8192 dwords

// ---------------------------------------------------------------------------
// TDM: 1-D dword-tile DMA global -> LDS. D# per CDNA5 ISA ch.8:
//  group0: [1:0]=count=1 | lds_addr | global_addr[31:0] | ga[56:32], type=2
//  group1: data_size=4B; tensor_dim0=dwords; tensor_dim1=1; tile_dim0=dwords
// ---------------------------------------------------------------------------
__device__ __forceinline__ void tdm_load_1d(unsigned lds_off, const void* gsrc,
                                            int dwords) {
    unsigned long long ga = (unsigned long long)(size_t)gsrc;
    u32x4 g0;
    g0[0] = 1u;                                        // count=1, user D#
    g0[1] = lds_off;                                   // lds_addr (bytes)
    g0[2] = (unsigned)ga;                              // global_addr[31:0]
    g0[3] = (unsigned)(ga >> 32) | (2u << 30);         // ga[56:32] | type=2
    int32x8 g1;
    g1[0] = (2 << 16);                                 // data_size = 4 bytes
    g1[1] = (dwords & 0xffff) << 16;                   // tensor_dim0[15:0]
    g1[2] = ((dwords >> 16) & 0xffff) | (1 << 16);     // td0[31:16] | tensor_dim1=1
    g1[3] = (dwords & 0xffff) << 16;                   // tile_dim0
    g1[4] = 0;                                         // tile_dim1/2 unused
    g1[5] = dwords;                                    // tensor_dim0_stride
    g1[6] = 0;
    g1[7] = 0;
    int32x4 z4 = {};
#if defined(__clang_major__) && __clang_major__ >= 23
    int32x8 z8 = {};
    __builtin_amdgcn_tensor_load_to_lds(g0, g1, z4, z4, z8, 0);
#else
    __builtin_amdgcn_tensor_load_to_lds(g0, g1, z4, z4, 0);
#endif
}

// ---------------------------------------------------------------------------
// Kernel 1: pack proj_left_w / proj_right_w (f32 [64][4096]) into native WMMA
// B-fragment layout, f16, fused as one N=128 matrix.
// B 32x16 frag: lane<16 -> K=0..7,16..23 of column n=lane&15; lane>=16 -> K=8..15,24..31.
// Layout: wpack[kt][nt][lane][16 halves]
// ---------------------------------------------------------------------------
__global__ void pack_w_kernel(const float* __restrict__ plw,
                              const float* __restrict__ prw,
                              _Float16* __restrict__ wpack) {
    int idx  = blockIdx.x * 256 + threadIdx.x;   // 128*8*32 = 32768
    int kt   = idx >> 8;
    int lane = idx & 31;
    int n    = ((idx >> 5) & 7) * 16 + (lane & 15);
    const float* wrow = (n < 64) ? (plw + (size_t)n * D_MODEL)
                                 : (prw + (size_t)(n - 64) * D_MODEL);
    int k0 = kt * 32 + ((lane & 16) ? 8 : 0);
    _Float16* dst = wpack + (size_t)idx * 16;
#pragma unroll
    for (int h = 0; h < 16; ++h) dst[h] = (_Float16)wrow[k0 + h + ((h >= 8) ? 8 : 0)];
}

// A 16x32 f16 fragment from a row of x (f32), converting on the fly.
// lane<16: K = {0..7,16..23}+0 ; lane>=16: +8 (kbase folded into pointer).
__device__ __forceinline__ v16h load_a_frag(const float* xp) {
    float4 f0 = *(const float4*)(xp + 0);
    float4 f1 = *(const float4*)(xp + 4);
    float4 f2 = *(const float4*)(xp + 16);
    float4 f3 = *(const float4*)(xp + 20);
    v16h a;
    a[0]=(_Float16)f0.x;  a[1]=(_Float16)f0.y;  a[2]=(_Float16)f0.z;  a[3]=(_Float16)f0.w;
    a[4]=(_Float16)f1.x;  a[5]=(_Float16)f1.y;  a[6]=(_Float16)f1.z;  a[7]=(_Float16)f1.w;
    a[8]=(_Float16)f2.x;  a[9]=(_Float16)f2.y;  a[10]=(_Float16)f2.z; a[11]=(_Float16)f2.w;
    a[12]=(_Float16)f3.x; a[13]=(_Float16)f3.y; a[14]=(_Float16)f3.z; a[15]=(_Float16)f3.w;
    return a;
}

// ---------------------------------------------------------------------------
// Kernel 2: fused left/right projection, M=32 rows per wave x N=128 strip.
// B fragments staged in LDS by the Tensor Data Mover (double buffered,
// overlapped with compute, s_wait_tensorcnt + barriers). Each B fragment
// feeds two v_wmma_f32_16x16x32_f16 (two M-halves) -> 512 MB total B reads.
// Epilogue adds bias, applies mask, stores f16 transposed [b][64][2048].
// ---------------------------------------------------------------------------
__global__ void __launch_bounds__(256)
proj_kernel(const float* __restrict__ x, const unsigned char* __restrict__ mask,
            const _Float16* __restrict__ wpack,
            const float* __restrict__ plb, const float* __restrict__ prb,
            _Float16* __restrict__ leftT, _Float16* __restrict__ rightT) {
    __shared__ _Float16 bsm[2 * CHUNK_HALVES];       // 64 KB of 320 KB/WGP

    int wave  = threadIdx.x >> 5;
    int lane  = threadIdx.x & 31;
    int row0  = (blockIdx.x * 8 + wave) * 32;        // 512 waves -> 64 blocks
    int kbase = (lane & 16) ? 8 : 0;
    const float* x0 = x + (size_t)(row0 +      (lane & 15)) * D_MODEL + kbase;
    const float* x1 = x + (size_t)(row0 + 16 + (lane & 15)) * D_MODEL + kbase;

    unsigned lds_base = (unsigned)(size_t)(&bsm[0]);

    v8f acc0[8] = {}, acc1[8] = {};

    // prime the pipeline: chunk 0 -> buffer 0
    if (threadIdx.x < 32) {
        tdm_load_1d(lds_base, wpack, CHUNK_DW);
        __builtin_amdgcn_s_wait_tensorcnt(0);
    }
    __syncthreads();

    for (int c = 0; c < NCHUNK; ++c) {
        // kick off next chunk's DMA while computing on this one
        if (threadIdx.x < 32 && (c + 1) < NCHUNK) {
            tdm_load_1d(lds_base + ((c + 1) & 1) * (CHUNK_HALVES * 2),
                        wpack + (size_t)(c + 1) * CHUNK_HALVES, CHUNK_DW);
        }
        const _Float16* lb = bsm + (c & 1) * CHUNK_HALVES;
#pragma unroll
        for (int ktl = 0; ktl < KC; ++ktl) {
            int kt = c * KC + ktl;
            const float* xp0 = x0 + kt * 32;
            const float* xp1 = x1 + kt * 32;
            __builtin_prefetch(xp0 + 512, 0, 1);
            __builtin_prefetch(xp1 + 512, 0, 1);
            v16h a0 = load_a_frag(xp0);
            v16h a1 = load_a_frag(xp1);
            const v16h* lbk = (const v16h*)lb + ktl * 256 + lane;
#pragma unroll
            for (int h = 0; h < 2; ++h) {
                v16h bf[4];
#pragma unroll
                for (int q = 0; q < 4; ++q) bf[q] = lbk[(h * 4 + q) * 32];
#pragma unroll
                for (int q = 0; q < 4; ++q) {
                    int nt = h * 4 + q;
                    acc0[nt] = __builtin_amdgcn_wmma_f32_16x16x32_f16(
                        false, a0, false, bf[q], (short)0, acc0[nt], false, false);
                    acc1[nt] = __builtin_amdgcn_wmma_f32_16x16x32_f16(
                        false, a1, false, bf[q], (short)0, acc1[nt], false, false);
                }
            }
        }
        __syncthreads();                      // everyone done reading buf[c&1]
        if ((c + 1) < NCHUNK) {
            if (threadIdx.x < 32) __builtin_amdgcn_s_wait_tensorcnt(0);
            __syncthreads();                  // buf[(c+1)&1] is ready
        }
    }

    // Epilogue: C/D layout -> rowg = row0 + g*16 + r + (lane>=16 ? 8 : 0)
    int half8 = (lane & 16) ? 8 : 0;
    int nlane = lane & 15;
#pragma unroll
    for (int g = 0; g < 2; ++g) {
        const v8f* acc = g ? acc1 : acc0;
        float mrowv[8];
#pragma unroll
        for (int r = 0; r < 8; ++r)
            mrowv[r] = mask[row0 + g * 16 + half8 + r] ? 1.0f : 0.0f;
#pragma unroll
        for (int nt = 0; nt < 8; ++nt) {
            int col = nt * 16 + nlane;
            float bias = (col < 64) ? plb[col] : prb[col - 64];
            _Float16* outT = (col < 64) ? leftT : rightT;
            int cc = (col < 64) ? col : (col - 64);
#pragma unroll
            for (int r = 0; r < 8; ++r) {
                int rowg = row0 + g * 16 + half8 + r;
                int b = rowg >> 11;
                int s = rowg & (SEQ - 1);
                float v = (acc[nt][r] + bias) * mrowv[r];
                outT[((size_t)b * 64 + cc) * SEQ + s] = (_Float16)v;
            }
        }
    }
}

// ---------------------------------------------------------------------------
// Kernel 3: cov[b] = left^T @ right / len + eps*I via WMMA over K=S=2048.
// One wave per 16x16 tile (8*4*4 = 128 waves); transposed f16 inputs make
// fragments contiguous loads.
// ---------------------------------------------------------------------------
__global__ void __launch_bounds__(32)
cov_kernel(const _Float16* __restrict__ leftT, const _Float16* __restrict__ rightT,
           const unsigned char* __restrict__ mask, float* __restrict__ cov) {
    int b    = blockIdx.x >> 4;
    int lt   = (blockIdx.x >> 2) & 3;
    int rt   = blockIdx.x & 3;
    int lane = threadIdx.x & 31;

    float cnt = 0.0f;
    const unsigned char* mb = mask + (size_t)b * SEQ;
    for (int i = lane; i < SEQ; i += 32) cnt += mb[i] ? 1.0f : 0.0f;
#pragma unroll
    for (int off = 16; off > 0; off >>= 1) cnt += __shfl_xor(cnt, off, 32);
    float invLen = 1.0f / fmaxf(cnt, 1.0f);

    int arow = lt * 16 + (lane & 15);
    int brow = rt * 16 + (lane & 15);
    const _Float16* ap = leftT  + ((size_t)b * 64 + arow) * SEQ + ((lane & 16) ? 16 : 0);
    const _Float16* bp = rightT + ((size_t)b * 64 + brow) * SEQ + ((lane & 16) ? 16 : 0);

    v8f acc = {};
    for (int s0 = 0; s0 < SEQ; s0 += 32) {
        v8h alo = *(const v8h*)(ap + s0);
        v8h ahi = *(const v8h*)(ap + s0 + 8);
        v16h a;
#pragma unroll
        for (int h = 0; h < 8; ++h) { a[h] = alo[h]; a[8 + h] = ahi[h]; }
        v16h bb = *(const v16h*)(bp + s0);
        acc = __builtin_amdgcn_wmma_f32_16x16x32_f16(
            false, a, false, bb, (short)0, acc, false, false);
    }
    int half8 = (lane & 16) ? 8 : 0;
#pragma unroll
    for (int r8 = 0; r8 < 8; ++r8) {
        int li = lt * 16 + half8 + r8;
        int ri = rt * 16 + (lane & 15);
        float v = acc[r8] * invLen;
        if (li == ri) v += EPSV;
        cov[((size_t)b * 64 + li) * 64 + ri] = v;
    }
}

// ---------------------------------------------------------------------------
// Kernel 4: batched Newton-Schulz sqrtm on 64x64, one workgroup per batch,
// everything in LDS.
// ---------------------------------------------------------------------------
__global__ void __launch_bounds__(256)
ns_sqrtm_kernel(const float* __restrict__ cov, float* __restrict__ out) {
    __shared__ float Y[64 * 64], Z[64 * 64], T[64 * 64];
    __shared__ float red[256];
    int b = blockIdx.x, t = threadIdx.x;
    const float* A = cov + (size_t)b * 4096;

    float ss = 0.0f;
    for (int e = t; e < 4096; e += 256) { float v = A[e]; ss += v * v; }
    red[t] = ss; __syncthreads();
    for (int off = 128; off > 0; off >>= 1) {
        if (t < off) red[t] += red[t + off];
        __syncthreads();
    }
    float norm = sqrtf(red[0]);
    float inv  = 1.0f / norm;
    for (int e = t; e < 4096; e += 256) {
        Y[e] = A[e] * inv;
        Z[e] = ((e >> 6) == (e & 63)) ? 1.0f : 0.0f;
    }
    __syncthreads();

    int row = t >> 2, j0 = (t & 3) << 4;
    for (int it = 0; it < 3; ++it) {
        float c[16];
#pragma unroll
        for (int j = 0; j < 16; ++j) c[j] = 0.0f;
        for (int k = 0; k < 64; ++k) {
            float a = Z[row * 64 + k];
#pragma unroll
            for (int j = 0; j < 16; ++j) c[j] += a * Y[k * 64 + j0 + j];
        }
#pragma unroll
        for (int j = 0; j < 16; ++j) {
            int col = j0 + j;
            T[row * 64 + col] = (row == col ? 1.5f : 0.0f) - 0.5f * c[j];
        }
        __syncthreads();
        float cy[16], cz[16];
#pragma unroll
        for (int j = 0; j < 16; ++j) { cy[j] = 0.0f; cz[j] = 0.0f; }
        for (int k = 0; k < 64; ++k) {
            float ay = Y[row * 64 + k];
            float at = T[row * 64 + k];
#pragma unroll
            for (int j = 0; j < 16; ++j) {
                cy[j] += ay * T[k * 64 + j0 + j];
                cz[j] += at * Z[k * 64 + j0 + j];
            }
        }
        __syncthreads();
#pragma unroll
        for (int j = 0; j < 16; ++j) {
            Y[row * 64 + j0 + j] = cy[j];
            Z[row * 64 + j0 + j] = cz[j];
        }
        __syncthreads();
    }
    float s = sqrtf(norm);
    for (int e = t; e < 4096; e += 256) out[(size_t)b * 4096 + e] = Y[e] * s;
}

// ---------------------------------------------------------------------------
// Kernel 5: hidden[b,n,h] = hl[n,h,:] @ covsqrt[b] @ hr[n,h,:]
// ---------------------------------------------------------------------------
__global__ void hidden_kernel(const float* __restrict__ covs,
                              const float* __restrict__ hlw,
                              const float* __restrict__ hrw,
                              float* __restrict__ hidden) {
    int idx = blockIdx.x * 256 + threadIdx.x;   // 8*384 = 3072
    int b = idx / 384, nh = idx % 384;
    const float* hlp = hlw + (size_t)nh * 64;
    const float* hrp = hrw + (size_t)nh * 64;
    const float* C   = covs + (size_t)b * 4096;
    float accv = 0.0f;
    for (int l = 0; l < 64; ++l) {
        const float* Crow = C + l * 64;
        float s = 0.0f;
#pragma unroll 8
        for (int r = 0; r < 64; ++r) s += Crow[r] * hrp[r];
        accv += hlp[l] * s;
    }
    hidden[idx] = accv;
}

// ---------------------------------------------------------------------------
// Kernel 6: output heads, concatenated [B, 10+100+1]
// ---------------------------------------------------------------------------
__global__ void out_kernel(const float* __restrict__ hidden,
                           const float* __restrict__ w0, const float* __restrict__ b0,
                           const float* __restrict__ w1, const float* __restrict__ b1,
                           const float* __restrict__ w2, const float* __restrict__ b2,
                           float* __restrict__ out) {
    int idx = blockIdx.x * 256 + threadIdx.x;
    if (idx >= BATCH * 111) return;
    int b = idx / 111, j = idx % 111;
    const float* w; float bias; int n;
    if (j < 10)       { w = w0 + (size_t)j * 128;        bias = b0[j];      n = 0; }
    else if (j < 110) { w = w1 + (size_t)(j - 10) * 128; bias = b1[j - 10]; n = 1; }
    else              { w = w2;                          bias = b2[0];      n = 2; }
    const float* h = hidden + ((size_t)b * 3 + n) * 128;
    float acc = bias;
    for (int k = 0; k < 128; ++k) acc += h[k] * w[k];
    out[idx] = acc;
}

// ---------------------------------------------------------------------------
extern "C" void kernel_launch(void* const* d_in, const int* in_sizes, int n_in,
                              void* d_out, int out_size, void* d_ws, size_t ws_size,
                              hipStream_t stream) {
    const float*         x    = (const float*)d_in[0];
    const unsigned char* mask = (const unsigned char*)d_in[1];  // jnp bool = 1 byte
    const float* plw = (const float*)d_in[2];
    const float* plb = (const float*)d_in[3];
    const float* prw = (const float*)d_in[4];
    const float* prb = (const float*)d_in[5];
    const float* hlw = (const float*)d_in[6];
    const float* hrw = (const float*)d_in[7];
    const float* w0  = (const float*)d_in[8];
    const float* b0  = (const float*)d_in[9];
    const float* w1  = (const float*)d_in[10];
    const float* b1  = (const float*)d_in[11];
    const float* w2  = (const float*)d_in[12];
    const float* b2  = (const float*)d_in[13];
    float* out = (float*)d_out;

    char* ws = (char*)d_ws;
    _Float16* wpack  = (_Float16*)(ws);                              // 1 MB
    _Float16* leftT  = (_Float16*)(ws + (1u << 20));                 // 2 MB
    _Float16* rightT = (_Float16*)(ws + (1u << 20) + (2u << 20));    // 2 MB
    float*    cov    = (float*)(ws + (5u << 20));                    // 128 KB
    float*    covs   = (float*)(ws + (5u << 20) + (128u << 10));     // 128 KB
    float*    hidden = (float*)(ws + (5u << 20) + (256u << 10));     // 12 KB

    pack_w_kernel<<<128, 256, 0, stream>>>(plw, prw, wpack);
    proj_kernel<<<64, 256, 0, stream>>>(x, mask, wpack, plb, prb, leftT, rightT);
    cov_kernel<<<128, 32, 0, stream>>>(leftT, rightT, mask, cov);
    ns_sqrtm_kernel<<<BATCH, 256, 0, stream>>>(cov, covs);
    hidden_kernel<<<12, 256, 0, stream>>>(covs, hlw, hrw, hidden);
    out_kernel<<<4, 256, 0, stream>>>(hidden, w0, b0, w1, b1, w2, b2, out);
}
